// GNN_COMP_51402168598782
// MI455X (gfx1250) — matrix operand order repaired and checked by reading, not compiled, not run on hardware
//
#include <hip/hip_runtime.h>
#include <hip/hip_bf16.h>

// ---------------------------------------------------------------------------
// Types for CDNA5 WMMA
// ---------------------------------------------------------------------------
typedef __bf16 bf16_t;
typedef bf16_t v16bf __attribute__((ext_vector_type(16)));
typedef bf16_t v8bf  __attribute__((ext_vector_type(8)));
typedef float  v8f   __attribute__((ext_vector_type(8)));

#define KDIM 128          // inner dimension for every GEMM in this net
#define HDIM 128          // hidden width
#define NEG_SLOPE 0.2f

// ---------------------------------------------------------------------------
// Small helpers
// ---------------------------------------------------------------------------
__device__ __forceinline__ unsigned ordf(float f) {
  unsigned u = __float_as_uint(f);
  return (u & 0x80000000u) ? ~u : (u | 0x80000000u);
}
__device__ __forceinline__ float deordf(unsigned u) {
  return (u & 0x80000000u) ? __uint_as_float(u & 0x7fffffffu)
                           : __uint_as_float(~u);
}
__device__ __forceinline__ void edge_sd(const int* __restrict__ ei, long long e,
                                        int E, int& s, int& d) {
  if (e < (long long)E) { s = ei[e]; d = ei[(long long)E + e]; }
  else                  { s = d = (int)(e - E); }   // self loop
}

// ---------------------------------------------------------------------------
// Fill / init kernels
// ---------------------------------------------------------------------------
__global__ void k_fill_f32(float* __restrict__ p, float v, long long count) {
  long long t = (long long)blockIdx.x * blockDim.x + threadIdx.x;
  if (t < count) p[t] = v;
}

__global__ void k_deg_accum(const int* __restrict__ ei, int E,
                            float* __restrict__ deg) {
  long long t = (long long)blockIdx.x * blockDim.x + threadIdx.x;
  if (t < E) atomicAdd(&deg[ei[(long long)E + t]], 1.0f);
}

__global__ void k_rsqrt_inplace(float* __restrict__ p, int n) {
  int t = blockIdx.x * blockDim.x + threadIdx.x;
  if (t < n) p[t] = rsqrtf(p[t]);
}

// ---------------------------------------------------------------------------
// WMMA GEMM:  Y[nrows x HOUT] = X[nrows x 128] @ W[128 x HOUT] (+ bias)
//
// 256 threads = 8 waves; each wave owns TWO 16-row tiles (32 rows); block
// covers 256 rows. B staged transposed in LDS (stride 136) so each lane's B
// fragment = 2x ds_load_b128. A fragments: unconditional float4 loads with a
// row *clamp* for the tail block (OOB A rows only feed OOB output rows, which
// are never stored) -- no per-element exec-mask predication. Full blocks take
// an unguarded store path; only the last block pays per-row guards.
// ---------------------------------------------------------------------------
template <int HOUT, bool HASBIAS>
__global__ void __launch_bounds__(256)
k_gemm_bf16(const float* __restrict__ X, const float* __restrict__ W,
            const float* __restrict__ bias, float* __restrict__ Y,
            int nrows) {
  constexpr int LDK = 136;                // pad: bank de-conflict, 16B aligned
  __shared__ bf16_t sBt[HOUT * LDK];      // transposed: sBt[n*LDK + k]

  const int tid = threadIdx.x;
  for (int idx = tid; idx < KDIM * HOUT; idx += 256) {
    const int k = idx / HOUT;
    const int n = idx % HOUT;
    sBt[n * LDK + k] = (bf16_t)W[idx];
  }
  __syncthreads();

  const int wave = tid >> 5;
  const int lane = tid & 31;
  const int m    = lane & 15;   // A row within tile / B column within tile
  const int half = lane >> 4;

  const long long rowBase   = (long long)blockIdx.x * 256 + wave * 32;
  const bool      fullBlock = ((long long)(blockIdx.x + 1) * 256 <= (long long)nrows);

  // Build A fragments for two row tiles x four K=32 chunks (reused across
  // every column tile). Unconditional vector loads; clamp row in tail block.
  v16bf afrag[2][4];
#pragma unroll
  for (int r = 0; r < 2; ++r) {
    long long row = rowBase + r * 16 + m;
    if (!fullBlock && row >= (long long)nrows) row = (long long)nrows - 1;
    const float* __restrict__ arow = X + row * KDIM;
#pragma unroll
    for (int kc = 0; kc < 4; ++kc) {
      const int kbase = kc * 32 + half * 8;
      float4 f0 = *(const float4*)(arow + kbase);
      float4 f1 = *(const float4*)(arow + kbase + 4);
      float4 f2 = *(const float4*)(arow + kbase + 16);
      float4 f3 = *(const float4*)(arow + kbase + 20);
      v16bf a;
      a[0]  = (bf16_t)f0.x; a[1]  = (bf16_t)f0.y;
      a[2]  = (bf16_t)f0.z; a[3]  = (bf16_t)f0.w;
      a[4]  = (bf16_t)f1.x; a[5]  = (bf16_t)f1.y;
      a[6]  = (bf16_t)f1.z; a[7]  = (bf16_t)f1.w;
      a[8]  = (bf16_t)f2.x; a[9]  = (bf16_t)f2.y;
      a[10] = (bf16_t)f2.z; a[11] = (bf16_t)f2.w;
      a[12] = (bf16_t)f3.x; a[13] = (bf16_t)f3.y;
      a[14] = (bf16_t)f3.z; a[15] = (bf16_t)f3.w;
      afrag[r][kc] = a;
    }
  }

#pragma unroll
  for (int t = 0; t < HOUT / 16; ++t) {
    const int n0 = t * 16;
    v8f c0 = {0.f, 0.f, 0.f, 0.f, 0.f, 0.f, 0.f, 0.f};
    v8f c1 = {0.f, 0.f, 0.f, 0.f, 0.f, 0.f, 0.f, 0.f};
#pragma unroll
    for (int kc = 0; kc < 4; ++kc) {
      const int kb = kc * 32 + half * 8;
      const bf16_t* fp = sBt + (n0 + m) * LDK + kb;   // 16B aligned
      v8bf lo = *(const v8bf*)(fp);
      v8bf hi = *(const v8bf*)(fp + 16);
      v16bf b = __builtin_shufflevector(lo, hi, 0, 1, 2, 3, 4, 5, 6, 7, 8, 9,
                                        10, 11, 12, 13, 14, 15);
      c0 = __builtin_amdgcn_wmma_f32_16x16x32_bf16(
               false, afrag[0][kc], false, b, (short)0, c0, false, false);
      c1 = __builtin_amdgcn_wmma_f32_16x16x32_bf16(
               false, afrag[1][kc], false, b, (short)0, c1, false, false);
    }
    const int   col  = n0 + m;
    const float badd = HASBIAS ? bias[col] : 0.0f;
    float* __restrict__ ybase = Y + rowBase * HOUT + col;
    if (fullBlock) {
#pragma unroll
      for (int i = 0; i < 8; ++i) {
        const int r0 = i + 8 * half;
        ybase[(long long)r0 * HOUT]        = c0[i] + badd;
        ybase[(long long)(r0 + 16) * HOUT] = c1[i] + badd;
      }
    } else {
#pragma unroll
      for (int i = 0; i < 8; ++i) {
        const int r0 = i + 8 * half;
        if (rowBase + r0 < (long long)nrows)
          ybase[(long long)r0 * HOUT] = c0[i] + badd;
        if (rowBase + r0 + 16 < (long long)nrows)
          ybase[(long long)(r0 + 16) * HOUT] = c1[i] + badd;
      }
    }
  }
}

// ---------------------------------------------------------------------------
// GCN scatter: wave per edge, 4 features per lane.
// agg[dst] += inv[src]*inv[dst] * xw[src]
// ---------------------------------------------------------------------------
__global__ void k_gcn_scatter(const float* __restrict__ xw,
                              const int* __restrict__ ei, int E, int n,
                              const float* __restrict__ inv,
                              float* __restrict__ agg) {
  long long t    = (long long)blockIdx.x * blockDim.x + threadIdx.x;
  long long edge = t >> 5;
  int lane       = (int)(t & 31);
  if (edge >= (long long)E + n) return;
  int s, d;
  edge_sd(ei, edge, E, s, d);
  float coef = inv[s] * inv[d];
  float4 v   = ((const float4*)xw)[(long long)s * 32 + lane];
  float* o   = agg + (long long)d * HDIM + lane * 4;
  atomicAdd(o + 0, coef * v.x);
  atomicAdd(o + 1, coef * v.y);
  atomicAdd(o + 2, coef * v.z);
  atomicAdd(o + 3, coef * v.w);
}

// ---------------------------------------------------------------------------
// Per-node attention dots: es[i] = xw[i].a_s ; ed[i] = xw[i].a_d
// Wave per node, wave32 shuffle reduction.
// ---------------------------------------------------------------------------
__global__ void k_att_dots(const float* __restrict__ xw,
                           const float* __restrict__ a_s,
                           const float* __restrict__ a_d,
                           float* __restrict__ es, float* __restrict__ ed,
                           int n) {
  long long t  = (long long)blockIdx.x * blockDim.x + threadIdx.x;
  long long nd = t >> 5;
  int lane     = (int)(t & 31);
  if (nd >= n) return;
  float4 p  = ((const float4*)xw)[nd * 32 + lane];
  float4 s4 = ((const float4*)a_s)[lane];
  float4 d4 = ((const float4*)a_d)[lane];
  float s = p.x * s4.x + p.y * s4.y + p.z * s4.z + p.w * s4.w;
  float d = p.x * d4.x + p.y * d4.y + p.z * d4.z + p.w * d4.w;
#pragma unroll
  for (int off = 16; off; off >>= 1) {
    s += __shfl_down(s, off, 32);
    d += __shfl_down(d, off, 32);
  }
  if (lane == 0) { es[nd] = s; ed[nd] = d; }
}

// ---------------------------------------------------------------------------
// GAT pass 1: e = leaky_relu(es[src]+ed[dst]); segment max via ordered-uint.
// (mb initialized to 0 == smallest ordered value; every dst has a self loop.)
// ---------------------------------------------------------------------------
__global__ void k_gat_logits(const float* __restrict__ es,
                             const float* __restrict__ ed,
                             const int* __restrict__ ei, int E, int n,
                             float* __restrict__ ebuf,
                             unsigned* __restrict__ mb) {
  long long t = (long long)blockIdx.x * blockDim.x + threadIdx.x;
  if (t >= (long long)E + n) return;
  int s, d;
  edge_sd(ei, t, E, s, d);
  float e = es[s] + ed[d];
  e = (e > 0.0f) ? e : NEG_SLOPE * e;
  ebuf[t] = e;
  atomicMax(&mb[d], ordf(e));
}

// GAT pass 2: ex = exp(e - max[dst]); denom[dst] += ex
__global__ void k_gat_exp(float* __restrict__ ebuf,
                          const int* __restrict__ ei, int E, int n,
                          const unsigned* __restrict__ mb,
                          float* __restrict__ denom) {
  long long t = (long long)blockIdx.x * blockDim.x + threadIdx.x;
  if (t >= (long long)E + n) return;
  int s, d;
  edge_sd(ei, t, E, s, d);
  float ex = expf(ebuf[t] - deordf(mb[d]));
  ebuf[t] = ex;
  atomicAdd(&denom[d], ex);
}

// GAT pass 3: agg[dst] += (ex/denom[dst]) * xw[src]; wave per edge.
__global__ void k_gat_scatter(const float* __restrict__ xw,
                              const float* __restrict__ ebuf,
                              const float* __restrict__ denom,
                              const int* __restrict__ ei, int E, int n,
                              float* __restrict__ agg) {
  long long t    = (long long)blockIdx.x * blockDim.x + threadIdx.x;
  long long edge = t >> 5;
  int lane       = (int)(t & 31);
  if (edge >= (long long)E + n) return;
  int s, d;
  edge_sd(ei, edge, E, s, d);
  float alpha = ebuf[edge] / denom[d];
  float4 v    = ((const float4*)xw)[(long long)s * 32 + lane];
  float* o    = agg + (long long)d * HDIM + lane * 4;
  atomicAdd(o + 0, alpha * v.x);
  atomicAdd(o + 1, alpha * v.y);
  atomicAdd(o + 2, alpha * v.z);
  atomicAdd(o + 3, alpha * v.w);
}

// ---------------------------------------------------------------------------
// Elementwise epilogue: out = relu(agg + bias[col]) over N*128
// ---------------------------------------------------------------------------
__global__ void k_bias_relu(const float* __restrict__ a,
                            const float* __restrict__ b,
                            float* __restrict__ o, long long total) {
  long long t = (long long)blockIdx.x * blockDim.x + threadIdx.x;
  if (t >= total) return;
  float v = a[t] + b[t & (HDIM - 1)];
  o[t] = (v > 0.0f) ? v : 0.0f;
}

// ---------------------------------------------------------------------------
// Host-side orchestration
// ---------------------------------------------------------------------------
static inline int blocks_for(long long work, int bs) {
  return (int)((work + bs - 1) / bs);
}

extern "C" void kernel_launch(void* const* d_in, const int* in_sizes, int n_in,
                              void* d_out, int out_size, void* d_ws,
                              size_t ws_size, hipStream_t stream) {
  const float* x     = (const float*)d_in[0];
  const int*   ei    = (const int*)d_in[1];
  const float* W1    = (const float*)d_in[2];
  const float* b1    = (const float*)d_in[3];
  const float* Wg1   = (const float*)d_in[4];
  const float* asrc1 = (const float*)d_in[5];
  const float* adst1 = (const float*)d_in[6];
  const float* bg1   = (const float*)d_in[7];
  const float* W2    = (const float*)d_in[8];
  const float* b2    = (const float*)d_in[9];
  const float* Wg2   = (const float*)d_in[10];
  const float* asrc2 = (const float*)d_in[11];
  const float* adst2 = (const float*)d_in[12];
  const float* bg2   = (const float*)d_in[13];
  const float* Wo    = (const float*)d_in[14];
  const float* bo    = (const float*)d_in[15];

  const int N  = in_sizes[0] / HDIM;        // 50000
  const int E  = in_sizes[1] / 2;           // 800000
  const long long ET = (long long)E + N;    // edges + self loops
  const long long NH = (long long)N * HDIM;

  // Workspace carve-out
  float*    p     = (float*)d_ws;
  float*    inv   = p;              p += N;       // degree -> 1/sqrt(deg)
  float*    es    = p;              p += N;
  float*    edv   = p;              p += N;
  float*    denom = p;              p += N;
  unsigned* mb    = (unsigned*)p;   p += N;
  float*    ebuf  = p;              p += ET;
  float*    xw    = p;              p += NH;
  float*    agg   = p;              p += NH;
  float*    h     = p;              p += NH;
  (void)ws_size; (void)n_in; (void)out_size;

  float* hout = (float*)d_out;                    // layer-4 activations (N x 128)
  float* zout = (float*)d_out + NH;               // logits (N x 64)

  const int BS = 256;
  const int gemmGrid = blocks_for(N, 256);        // 256 rows per block

  // --- GCN normalization coefficients ------------------------------------
  k_fill_f32<<<blocks_for(N, BS), BS, 0, stream>>>(inv, 1.0f, N);   // self loop
  k_deg_accum<<<blocks_for(E, BS), BS, 0, stream>>>(ei, E, inv);
  k_rsqrt_inplace<<<blocks_for(N, BS), BS, 0, stream>>>(inv, N);

  // --- Layer 1: GCN(x, W1, b1) + ReLU -> h --------------------------------
  k_gemm_bf16<HDIM, false><<<gemmGrid, BS, 0, stream>>>(x, W1, nullptr, xw, N);
  k_fill_f32<<<blocks_for(NH, BS), BS, 0, stream>>>(agg, 0.0f, NH);
  k_gcn_scatter<<<blocks_for(ET * 32, BS), BS, 0, stream>>>(xw, ei, E, N, inv, agg);
  k_bias_relu<<<blocks_for(NH, BS), BS, 0, stream>>>(agg, b1, h, NH);

  // --- Layer 2: GAT(h, Wg1, asrc1, adst1, bg1) + ReLU -> h ----------------
  k_gemm_bf16<HDIM, false><<<gemmGrid, BS, 0, stream>>>(h, Wg1, nullptr, xw, N);
  k_att_dots<<<blocks_for((long long)N * 32, BS), BS, 0, stream>>>(xw, asrc1, adst1, es, edv, N);
  k_fill_f32<<<blocks_for(N, BS), BS, 0, stream>>>((float*)mb, 0.0f, N);
  k_fill_f32<<<blocks_for(N, BS), BS, 0, stream>>>(denom, 0.0f, N);
  k_fill_f32<<<blocks_for(NH, BS), BS, 0, stream>>>(agg, 0.0f, NH);
  k_gat_logits<<<blocks_for(ET, BS), BS, 0, stream>>>(es, edv, ei, E, N, ebuf, mb);
  k_gat_exp<<<blocks_for(ET, BS), BS, 0, stream>>>(ebuf, ei, E, N, mb, denom);
  k_gat_scatter<<<blocks_for(ET * 32, BS), BS, 0, stream>>>(xw, ebuf, denom, ei, E, N, agg);
  k_bias_relu<<<blocks_for(NH, BS), BS, 0, stream>>>(agg, bg1, h, NH);

  // --- Layer 3: GCN(h, W2, b2) + ReLU -> h --------------------------------
  k_gemm_bf16<HDIM, false><<<gemmGrid, BS, 0, stream>>>(h, W2, nullptr, xw, N);
  k_fill_f32<<<blocks_for(NH, BS), BS, 0, stream>>>(agg, 0.0f, NH);
  k_gcn_scatter<<<blocks_for(ET * 32, BS), BS, 0, stream>>>(xw, ei, E, N, inv, agg);
  k_bias_relu<<<blocks_for(NH, BS), BS, 0, stream>>>(agg, b2, h, NH);

  // --- Layer 4: GAT(h, Wg2, asrc2, adst2, bg2) + ReLU -> d_out[0:N*H] -----
  k_gemm_bf16<HDIM, false><<<gemmGrid, BS, 0, stream>>>(h, Wg2, nullptr, xw, N);
  k_att_dots<<<blocks_for((long long)N * 32, BS), BS, 0, stream>>>(xw, asrc2, adst2, es, edv, N);
  k_fill_f32<<<blocks_for(N, BS), BS, 0, stream>>>((float*)mb, 0.0f, N);
  k_fill_f32<<<blocks_for(N, BS), BS, 0, stream>>>(denom, 0.0f, N);
  k_fill_f32<<<blocks_for(NH, BS), BS, 0, stream>>>(agg, 0.0f, NH);
  k_gat_logits<<<blocks_for(ET, BS), BS, 0, stream>>>(es, edv, ei, E, N, ebuf, mb);
  k_gat_exp<<<blocks_for(ET, BS), BS, 0, stream>>>(ebuf, ei, E, N, mb, denom);
  k_gat_scatter<<<blocks_for(ET * 32, BS), BS, 0, stream>>>(xw, ebuf, denom, ei, E, N, agg);
  k_bias_relu<<<blocks_for(NH, BS), BS, 0, stream>>>(agg, bg2, hout, NH);

  // --- Output head: z = h @ Wo + bo -> d_out[N*H : N*H + N*C] -------------
  k_gemm_bf16<64, true><<<gemmGrid, BS, 0, stream>>>(hout, Wo, bo, zout, N);
}
